// FNO2d_81338090652051
// MI455X (gfx1250) — matrix operand order repaired and checked by reading, not compiled, not run on hardware
//
#include <hip/hip_runtime.h>
#include <hip/hip_bf16.h>
#include <math.h>

// ---------------------------------------------------------------------------
// FNO2d forward for MI455X (gfx1250, wave32, WMMA, TDM).
//  - activations f16 channels-last [pixel][64], fp32 accumulation
//  - truncated-mode spectral conv as separable DFT GEMMs, all on WMMA:
//      conv1x1/fc1 : [128pix x 64] x [64 x 64]          (k_chan_gemm)
//      DFT over x  : [32 twiddle x 256] x [256 x 64ci]  (k_dft1, TDM-fed)
//      inv over x  : [256x x 32] x [32 x 64co] + fuse   (k_inv2, TDM-fed)
//  - small steps (dft2, mix, inv1) stay fp32 VALU for precision
// ---------------------------------------------------------------------------

typedef __attribute__((ext_vector_type(16))) _Float16 v16h;
typedef __attribute__((ext_vector_type(8)))  float    v8f;
typedef __attribute__((ext_vector_type(4)))  unsigned int u32x4;
typedef __attribute__((ext_vector_type(4)))  int          i32x4;
typedef __attribute__((ext_vector_type(8)))  int          i32x8;
typedef _Float16 h16_t;

#define B_    16
#define CIN_  3
#define N_    256
#define W_    64
#define M_    16
#define NP    65536              /* N_*N_ */
#define P_TOT (B_ * NP)          /* 1,048,576 pixels */

__device__ __forceinline__ float gelu_exact(float x) {
    return 0.5f * x * (1.0f + erff(x * 0.70710678118654752f));
}

// ---------------------------------------------------------------------------
// Tensor Data Mover: 2D tile load Global->LDS (data_size=2B lines).
// D# bit layout per CDNA5 ISA sec 8.3 (group0) / 8.4 (group1).
// ---------------------------------------------------------------------------
__device__ __forceinline__ void tdm_load_2d_f16(unsigned lds_off, const void* gptr,
                                                unsigned tile_d0, unsigned tile_d1,
                                                unsigned long long stride0) {
    unsigned long long ga = (unsigned long long)(uintptr_t)gptr;
    u32x4 g0;
    g0.x = 1u;                                   // count=1, user mode, no gather
    g0.y = lds_off;                              // lds_addr
    g0.z = (unsigned)(ga & 0xFFFFFFFFu);         // global_addr[31:0]
    g0.w = (unsigned)((ga >> 32) & 0x01FFFFFFu)  // global_addr[56:32]
         | (2u << 30);                           // type=2 ("image")
    i32x8 g1;
    g1[0] = (int)(1u << 16);                     // data_size=1 (2 bytes)
    g1[1] = (int)((tile_d0 & 0xFFFFu) << 16);    // tensor_dim0[15:0] (== tile_d0)
    g1[2] = (int)(((tile_d0 >> 16) & 0xFFFFu) | ((tile_d1 & 0xFFFFu) << 16)); // tensor_dim1
    g1[3] = (int)(((tile_d1 >> 16) & 0xFFFFu) | ((tile_d0 & 0xFFFFu) << 16)); // tile_dim0
    g1[4] = (int)(tile_d1 & 0xFFFFu);            // tile_dim1 ; tile_dim2=0
    g1[5] = (int)(stride0 & 0xFFFFFFFFull);      // tensor_dim0_stride[31:0]
    g1[6] = (int)((stride0 >> 32) & 0xFFFFull);  // tensor_dim0_stride[47:32]
    g1[7] = 0;
    i32x4 gz = {0, 0, 0, 0};
#if __clang_major__ >= 23
    i32x8 gz8 = {0, 0, 0, 0, 0, 0, 0, 0};
    __builtin_amdgcn_tensor_load_to_lds(g0, g1, gz, gz, gz8, 0);
#else
    __builtin_amdgcn_tensor_load_to_lds(g0, g1, gz, gz, 0);
#endif
}

// ---------------------------------------------------------------------------
// fc0 lift: h16[p][w] = sum_c x[b,c,:,:] * fc0_w[w,c] + fc0_b[w]   (K=3, VALU)
// ---------------------------------------------------------------------------
__global__ void __launch_bounds__(256)
k_fc0(const float* __restrict__ x, const float* __restrict__ w,
      const float* __restrict__ b, h16_t* __restrict__ h16) {
    __shared__ float swt[W_ * CIN_];
    __shared__ float sb[W_];
    int t = threadIdx.x;
    if (t < W_ * CIN_) swt[t] = w[t];
    if (t < W_) sb[t] = b[t];
    __syncthreads();
    int p = blockIdx.x * 256 + t;
    int bb = p >> 16;
    int rem = p & (NP - 1);
    float xv[CIN_];
#pragma unroll
    for (int c = 0; c < CIN_; ++c)
        xv[c] = x[(size_t)(bb * CIN_ + c) * NP + rem];
    unsigned* dst = (unsigned*)(h16 + (size_t)p * W_);
#pragma unroll 8
    for (int wo = 0; wo < W_; wo += 2) {
        float a0 = sb[wo], a1 = sb[wo + 1];
#pragma unroll
        for (int c = 0; c < CIN_; ++c) {
            a0 = fmaf(xv[c], swt[wo * CIN_ + c], a0);
            a1 = fmaf(xv[c], swt[(wo + 1) * CIN_ + c], a1);
        }
        union { h16_t h[2]; unsigned u; } pk;
        pk.h[0] = (h16_t)a0; pk.h[1] = (h16_t)a1;
        dst[wo >> 1] = pk.u;
    }
}

// ---------------------------------------------------------------------------
// 64x64 channel GEMM on WMMA (conv1x1 / fc1).
// ---------------------------------------------------------------------------
__global__ void __launch_bounds__(256)
k_chan_gemm(const h16_t* __restrict__ hin, const float* __restrict__ w,
            const float* __restrict__ bias, h16_t* __restrict__ out, int doGelu) {
    __shared__ __align__(16) h16_t wfrag[2 * 4 * 32 * 16]; // 8 KB, B-fragment order
    __shared__ __align__(16) h16_t sout[128 * W_];         // 16 KB store staging
    __shared__ float sb[W_];
    int t = threadIdx.x;
    int blockPix = blockIdx.x * 128;

    for (int i = t; i < 4096; i += 256) {
        int e  = i & 15;
        int ln = (i >> 4) & 31;
        int ct = (i >> 9) & 3;
        int kc = i >> 11;
        int ci = kc * 32 + ((ln >> 4) << 3) + (e < 8 ? e : e + 8);
        int co = ct * 16 + (ln & 15);
        wfrag[i] = (h16_t)w[co * W_ + ci];
    }
    if (t < W_) sb[t] = bias[t];
    if (blockIdx.x + 1 < gridDim.x)
        __builtin_prefetch(hin + (size_t)(blockPix + 128) * W_ + (size_t)t * 4, 0, 1);
    __syncthreads();

    int lane = t & 31;
    int wv   = t >> 5;
    int m    = lane & 15;
    int hi   = lane >> 4;
    int wavePix = wv * 16;

    const uint4* arow = (const uint4*)(hin + (size_t)(blockPix + wavePix + m) * W_);

    v8f acc[4] = {};
#pragma unroll
    for (int kc = 0; kc < 2; ++kc) {
        union { v16h v; uint4 q[2]; } af;
        af.q[0] = arow[kc * 4 + hi];
        af.q[1] = arow[kc * 4 + hi + 2];
#pragma unroll
        for (int ct = 0; ct < 4; ++ct) {
            union { v16h v; uint4 q[2]; } bf;
            const uint4* bp = (const uint4*)&wfrag[((kc * 4 + ct) * 32 + lane) * 16];
            bf.q[0] = bp[0];
            bf.q[1] = bp[1];
            acc[ct] = __builtin_amdgcn_wmma_f32_16x16x32_f16(
                false, af.v, false, bf.v, (short)0, acc[ct], false, false);
        }
    }
#pragma unroll
    for (int ct = 0; ct < 4; ++ct) {
#pragma unroll
        for (int r = 0; r < 8; ++r) {
            int M  = hi ? (r + 8) : r;
            int co = ct * 16 + m;
            float v = acc[ct][r] + sb[co];
            if (doGelu) v = gelu_exact(v);
            sout[(wavePix + M) * W_ + co] = (h16_t)v;
        }
    }
    __syncthreads();
    {
        uint4* g = (uint4*)(out + (size_t)blockPix * W_);
        const uint4* s = (const uint4*)sout;
        for (int i = t; i < 1024; i += 256) g[i] = s[i];
    }
}

// ---------------------------------------------------------------------------
// Forward DFT over x on WMMA:  E[32 x 256] (16 cos rows, 16 -sin rows, f16)
//                            x h[256 x 64ci]  ->  t1[b][y][32][64] fp32
// grid (y, b). TDM loads the [256][64] f16 x-column tile; B fragments staged
// in two 4-K-chunk passes (LDS <= ~58 KB); A built per-lane from sincos table.
// ---------------------------------------------------------------------------
__global__ void __launch_bounds__(256)
k_dft1(const h16_t* __restrict__ h16, float* __restrict__ t1) {
    __shared__ __align__(16) h16_t sh[N_ * W_];        // 32 KB [x][ci]
    __shared__ __align__(16) h16_t bfrag[4 * 4 * 32 * 16]; // 16 KB B-fragment order
    __shared__ float sout[32 * W_];                    // 8 KB result staging
    __shared__ float tc[256], tsn[256];
    int t = threadIdx.x;
    int y = blockIdx.x, b = blockIdx.y;

    if (t < 32) {
        tdm_load_2d_f16((unsigned)(uintptr_t)(void*)sh,
                        h16 + ((size_t)(b * N_) * N_ + y) * W_,
                        W_, N_, (unsigned long long)N_ * W_);
    }
    {
        float s, c;
        sincosf(6.28318530717958648f * (float)t / 256.0f, &s, &c);
        tc[t] = c; tsn[t] = s;
    }
    if (t < 32) __builtin_amdgcn_s_wait_tensorcnt(0);
    __syncthreads();

    int lane = t & 31;
    int wv   = t >> 5;
    int m    = lane & 15;
    int hi   = lane >> 4;
    int mt   = wv >> 2;          // 0: cos rows, 1: -sin rows
    int nt   = wv & 3;           // ci tile

    v8f acc = {};
#pragma unroll
    for (int P = 0; P < 2; ++P) {
        // stage B fragments for K chunks P*4 .. P*4+3
        for (int i = t; i < 8192; i += 256) {
            int e   = i & 15;
            int ln  = (i >> 4) & 31;
            int kcl = (i >> 9) & 3;
            int ntt = i >> 11;
            int K   = (P * 4 + kcl) * 32 + ((ln >> 4) << 4) + e;
            int ci  = ntt * 16 + (ln & 15);
            bfrag[i] = sh[K * W_ + ci];
        }
        __syncthreads();
#pragma unroll
        for (int kcl = 0; kcl < 4; ++kcl) {
            int kc = P * 4 + kcl;
            union { v16h v; h16_t h[16]; } af;
#pragma unroll
            for (int e = 0; e < 16; ++e) {
                int k = kc * 32 + (hi ? 16 : 0) + e;
                float tv = (mt == 0) ? tc[(m * k) & 255] : -tsn[(m * k) & 255];
                af.h[e] = (h16_t)tv;
            }
            union { v16h v; uint4 q[2]; } bf;
            const uint4* bp = (const uint4*)&bfrag[((nt * 4 + kcl) * 32 + lane) * 16];
            bf.q[0] = bp[0];
            bf.q[1] = bp[1];
            acc = __builtin_amdgcn_wmma_f32_16x16x32_f16(
                false, af.v, false, bf.v, (short)0, acc, false, false);
        }
        __syncthreads();
    }
    const float nrm = 1.0f / 256.0f;   // ortho forward
#pragma unroll
    for (int rr = 0; rr < 8; ++rr) {
        int R = mt * 16 + (hi ? rr + 8 : rr);
        sout[R * W_ + nt * 16 + m] = acc[rr] * nrm;
    }
    __syncthreads();
    size_t base = (size_t)(b * N_ + y) * (32 * W_);
    for (int i = t; i < 32 * W_; i += 256) t1[base + i] = sout[i];
}

// ---------------------------------------------------------------------------
// Forward DFT over y (K=256): Xft[b][k1][k2][ci] = sum_y t1 * e^{-2pi i k2 y/256}
// t1 layout [b][y][32][64]: re row = k1, im row = 16+k1.
// ---------------------------------------------------------------------------
__global__ void __launch_bounds__(256)
k_dft2(const float* __restrict__ t1,
       float* __restrict__ Xre, float* __restrict__ Xim) {
    __shared__ float sr[W_ * 64], si[W_ * 64];    // 16 KB each, [ci][yy]
    __shared__ float tc[256], tsn[256];
    int t = threadIdx.x;
    int k1 = blockIdx.x, b = blockIdx.y;
    { float s, c; sincosf(6.28318530717958648f * (float)t / 256.0f, &s, &c); tc[t] = c; tsn[t] = s; }
    int k2  = t >> 4;
    int ci0 = (t & 15) * 4;
    float re[4] = {}, im[4] = {};
    for (int cch = 0; cch < 4; ++cch) {
        __syncthreads();
        for (int i = t; i < W_ * 64; i += 256) {
            int ci = i & 63, yy = i >> 6;
            size_t gb2 = (size_t)(b * N_ + cch * 64 + yy) * (32 * W_);
            sr[ci * 64 + yy] = t1[gb2 + k1 * W_ + ci];
            si[ci * 64 + yy] = t1[gb2 + (M_ + k1) * W_ + ci];
        }
        __syncthreads();
        for (int yy = 0; yy < 64; ++yy) {
            int yv  = cch * 64 + yy;
            int idx = (k2 * yv) & 255;
            float c = tc[idx], s = tsn[idx];
#pragma unroll
            for (int j = 0; j < 4; ++j) {
                float tr = sr[(ci0 + j) * 64 + yy];
                float ti = si[(ci0 + j) * 64 + yy];
                re[j] += tr * c + ti * s;      // (tr+i ti)(c - i s)
                im[j] += ti * c - tr * s;
            }
        }
    }
    size_t ob = (size_t)((b * M_ + k1) * M_ + k2) * W_ + ci0;
#pragma unroll
    for (int j = 0; j < 4; ++j) { Xre[ob + j] = re[j]; Xim[ob + j] = im[j]; }
}

// ---------------------------------------------------------------------------
// Mode mix: per-(k1,k2) complex 64x64 channel GEMM over all 16 batches.
// ---------------------------------------------------------------------------
__global__ void __launch_bounds__(256)
k_mix(const float* __restrict__ Xre, const float* __restrict__ Xim,
      const float* __restrict__ sw, float* __restrict__ Yre, float* __restrict__ Yim) {
    __shared__ float wr[W_ * W_], wi[W_ * W_];   // 16 KB each [ci][co]
    __shared__ float xr[B_ * W_], xi[B_ * W_];   //  4 KB each [b][ci]
    int t = threadIdx.x;
    int k2 = blockIdx.x, k1 = blockIdx.y;
    int m2 = (k1 * M_ + k2) * 2;
    for (int i = t; i < W_ * W_; i += 256) {      // sw: [ci][co][16][16][2]
        wr[i] = sw[(size_t)i * 512 + m2];
        wi[i] = sw[(size_t)i * 512 + m2 + 1];
    }
    for (int i = t; i < B_ * W_; i += 256) {
        int bb = i >> 6, ci = i & 63;
        size_t g = (size_t)((bb * M_ + k1) * M_ + k2) * W_ + ci;
        xr[i] = Xre[g]; xi[i] = Xim[g];
    }
    __syncthreads();
    int bb = t >> 4, co0 = (t & 15) * 4;
    float yr[4] = {}, yi[4] = {};
    for (int ci = 0; ci < W_; ++ci) {
        float ar = xr[bb * W_ + ci], ai = xi[bb * W_ + ci];
#pragma unroll
        for (int j = 0; j < 4; ++j) {
            float br = wr[ci * W_ + co0 + j], bi = wi[ci * W_ + co0 + j];
            yr[j] += ar * br - ai * bi;
            yi[j] += ar * bi + ai * br;
        }
    }
    size_t ob = (size_t)((bb * M_ + k1) * M_ + k2) * W_ + co0;
#pragma unroll
    for (int j = 0; j < 4; ++j) { Yre[ob + j] = yr[j]; Yim[ob + j] = yi[j]; }
}

// ---------------------------------------------------------------------------
// Inverse over k2 (Hermitian weights: 1 for k2=0, 2 otherwise):
// t2[b][y][k1][co]{re,im} = sum_k2 wgt(k2) * Yft * e^{+2pi i k2 y/256}
// ---------------------------------------------------------------------------
__global__ void __launch_bounds__(256)
k_inv1(const float* __restrict__ Yre, const float* __restrict__ Yim, float* __restrict__ t2) {
    __shared__ float sYr[M_ * W_], sYi[M_ * W_];  // 4 KB each
    __shared__ float tc[256], tsn[256];
    int t = threadIdx.x;
    int k1 = blockIdx.x, b = blockIdx.y;
    { float s, c; sincosf(6.28318530717958648f * (float)t / 256.0f, &s, &c); tc[t] = c; tsn[t] = s; }
    size_t gb = (size_t)(b * M_ + k1) * M_ * W_;
    for (int i = t; i < M_ * W_; i += 256) { sYr[i] = Yre[gb + i]; sYi[i] = Yim[gb + i]; }
    __syncthreads();
    int y = t;
    float wc_[M_], ws_[M_];
#pragma unroll
    for (int k2 = 0; k2 < M_; ++k2) {
        int idx = (k2 * y) & 255;
        float g = (k2 == 0) ? 1.0f : 2.0f;
        wc_[k2] = g * tc[idx];
        ws_[k2] = g * tsn[idx];
    }
    size_t ob = ((size_t)(b * N_ + y) * M_ + k1) * (W_ * 2);
    for (int co = 0; co < W_; ++co) {
        float re = 0.f, im = 0.f;
#pragma unroll
        for (int k2 = 0; k2 < M_; ++k2) {
            float ar = sYr[k2 * W_ + co], ai = sYi[k2 * W_ + co];
            re += ar * wc_[k2] - ai * ws_[k2];   // (ar+i ai)(c + i s)
            im += ar * ws_[k2] + ai * wc_[k2];
        }
        t2[ob + co * 2]     = re;
        t2[ob + co * 2 + 1] = im;
    }
}

// ---------------------------------------------------------------------------
// Inverse over k1 on WMMA + fusion:
//   Einv[256x x 32] (cos | -sin, f16) x t2c[32 x 64co] -> [256x x 64co]
//   out = gelu(result/256 + conv1x1)   written back to h16.
// grid (y, b). conv tile arrives by TDM; its LDS tile is overwritten in place
// with the fused f16 result, then stored as coalesced 128B lines.
// ---------------------------------------------------------------------------
__global__ void __launch_bounds__(256)
k_inv2(const float* __restrict__ t2, const h16_t* __restrict__ yc16,
       h16_t* __restrict__ h16) {
    __shared__ __align__(16) h16_t yio[N_ * W_];       // 32 KB conv tile / result
    __shared__ __align__(16) h16_t bfrag[4 * 32 * 16]; // 4 KB t2 in B-fragment order
    __shared__ float tc[256], tsn[256];
    int t = threadIdx.x;
    int y = blockIdx.x, b = blockIdx.y;

    if (t < 32) {
        tdm_load_2d_f16((unsigned)(uintptr_t)(void*)yio,
                        yc16 + ((size_t)(b * N_) * N_ + y) * W_,
                        W_, N_, (unsigned long long)N_ * W_);
    }
    {
        float s, c;
        sincosf(6.28318530717958648f * (float)t / 256.0f, &s, &c);
        tc[t] = c; tsn[t] = s;
    }
    // t2 -> B fragment order: row K<16 = re(k1=K), K>=16 = im(k1=K-16)
    size_t gb = (size_t)(b * N_ + y) * (M_ * W_ * 2);
    for (int i = t; i < 2048; i += 256) {
        int e  = i & 15;
        int ln = (i >> 4) & 31;
        int ct = i >> 9;
        int K  = (ln >= 16 ? 16 : 0) + e;
        int k1 = K & 15, ri = K >> 4;
        int co = ct * 16 + (ln & 15);
        bfrag[i] = (h16_t)t2[gb + k1 * 128 + co * 2 + ri];
    }
    if (t < 32) __builtin_amdgcn_s_wait_tensorcnt(0);
    __syncthreads();

    int lane = t & 31;
    int wv   = t >> 5;
    int m    = lane & 15;
    int hi   = lane >> 4;

    union { v16h v; uint4 q[2]; } bf[4];
#pragma unroll
    for (int ct = 0; ct < 4; ++ct) {
        const uint4* bp = (const uint4*)&bfrag[(ct * 32 + lane) * 16];
        bf[ct].q[0] = bp[0];
        bf[ct].q[1] = bp[1];
    }
    v8f acc[2][4] = {};
#pragma unroll
    for (int mt2 = 0; mt2 < 2; ++mt2) {
        int r = (2 * wv + mt2) * 16 + m;           // x row
        union { v16h v; h16_t h[16]; } af;         // A[x][k]: cos | -sin
#pragma unroll
        for (int e = 0; e < 16; ++e) {
            int k = (hi ? 16 : 0) + e;
            float tv = (k < 16) ? tc[(r * k) & 255] : -tsn[(r * (k - 16)) & 255];
            af.h[e] = (h16_t)tv;
        }
#pragma unroll
        for (int ct = 0; ct < 4; ++ct)
            acc[mt2][ct] = __builtin_amdgcn_wmma_f32_16x16x32_f16(
                false, af.v, false, bf[ct].v, (short)0, acc[mt2][ct], false, false);
    }
    const float nrm = 1.0f / 256.0f;   // ortho inverse
#pragma unroll
    for (int mt2 = 0; mt2 < 2; ++mt2) {
#pragma unroll
        for (int ct = 0; ct < 4; ++ct) {
#pragma unroll
            for (int rr = 0; rr < 8; ++rr) {
                int R  = (2 * wv + mt2) * 16 + (hi ? rr + 8 : rr);
                int co = ct * 16 + m;
                float v = acc[mt2][ct][rr] * nrm + (float)yio[R * W_ + co];
                yio[R * W_ + co] = (h16_t)gelu_exact(v);
            }
        }
    }
    __syncthreads();
    for (int i = t; i < 2048; i += 256) {
        int row = i >> 3, q = i & 7;
        uint4* dst = (uint4*)(h16 + (((size_t)(b * N_ + row)) * N_ + y) * W_);
        dst[q] = ((const uint4*)yio)[i];
    }
}

// ---------------------------------------------------------------------------
// fc2 head: out[p] = sum_f g16[p][f]*fc2_w[f] + fc2_b
// ---------------------------------------------------------------------------
__global__ void __launch_bounds__(256)
k_fc2(const h16_t* __restrict__ g16, const float* __restrict__ w,
      const float* __restrict__ b, float* __restrict__ out) {
    __shared__ float swt[W_];
    int t = threadIdx.x;
    if (t < W_) swt[t] = w[t];
    __syncthreads();
    int p = blockIdx.x * 256 + t;
    size_t base = (size_t)p * W_;
    float a = b[0];
#pragma unroll 8
    for (int f = 0; f < W_; ++f) a = fmaf((float)g16[base + f], swt[f], a);
    out[p] = a;
}

// ---------------------------------------------------------------------------
extern "C" void kernel_launch(void* const* d_in, const int* in_sizes, int n_in,
                              void* d_out, int out_size, void* d_ws, size_t ws_size,
                              hipStream_t stream) {
    (void)in_sizes; (void)n_in; (void)out_size; (void)ws_size;
    const float* x     = (const float*)d_in[0];
    const float* fc0w  = (const float*)d_in[1];
    const float* fc0b  = (const float*)d_in[2];
    const float* sws[4] = {(const float*)d_in[3], (const float*)d_in[4],
                           (const float*)d_in[5], (const float*)d_in[6]};
    const float* cws[4] = {(const float*)d_in[7],  (const float*)d_in[9],
                           (const float*)d_in[11], (const float*)d_in[13]};
    const float* cbs[4] = {(const float*)d_in[8],  (const float*)d_in[10],
                           (const float*)d_in[12], (const float*)d_in[14]};
    const float* fc1w = (const float*)d_in[15];
    const float* fc1b = (const float*)d_in[16];
    const float* fc2w = (const float*)d_in[17];
    const float* fc2b = (const float*)d_in[18];

    // workspace layout (~324 MB total)
    char*  ws   = (char*)d_ws;
    h16_t* h16  = (h16_t*)ws;                                   // 128 MB
    h16_t* yc16 = (h16_t*)(ws + (size_t)P_TOT * W_ * 2);        // 128 MB (also fc1 out)
    float* t1   = (float*)(ws + (size_t)P_TOT * W_ * 4);        // 32 MB [b][y][32][64]
    float* t2   = t1 + (size_t)B_ * N_ * 32 * W_;               // 32 MB
    float* Xre  = t2 + (size_t)B_ * N_ * M_ * W_ * 2;           // 1 MB
    float* Xim  = Xre + (size_t)B_ * M_ * M_ * W_;
    float* Yre  = Xim + (size_t)B_ * M_ * M_ * W_;
    float* Yim  = Yre + (size_t)B_ * M_ * M_ * W_;

    k_fc0<<<P_TOT / 256, 256, 0, stream>>>(x, fc0w, fc0b, h16);
    for (int L = 0; L < 4; ++L) {
        k_chan_gemm<<<P_TOT / 128, 256, 0, stream>>>(h16, cws[L], cbs[L], yc16, 0);
        k_dft1<<<dim3(N_, B_), 256, 0, stream>>>(h16, t1);
        k_dft2<<<dim3(M_, B_), 256, 0, stream>>>(t1, Xre, Xim);
        k_mix <<<dim3(M_, M_), 256, 0, stream>>>(Xre, Xim, sws[L], Yre, Yim);
        k_inv1<<<dim3(M_, B_), 256, 0, stream>>>(Yre, Yim, t2);
        k_inv2<<<dim3(N_, B_), 256, 0, stream>>>(t2, yc16, h16);
    }
    k_chan_gemm<<<P_TOT / 128, 256, 0, stream>>>(h16, fc1w, fc1b, yc16, 1);
    k_fc2<<<P_TOT / 256, 256, 0, stream>>>(yc16, fc2w, fc2b, (float*)d_out);
}
